// GCNLinkPredictor_3212635537794
// MI455X (gfx1250) — matrix operand order repaired and checked by reading, not compiled, not run on hardware
//
#include <hip/hip_runtime.h>
#include <hip/hip_bf16.h>

typedef __attribute__((ext_vector_type(16))) _Float16 v16h;
typedef __attribute__((ext_vector_type(8)))  _Float16 v8h;
typedef __attribute__((ext_vector_type(8)))  float    v8f;
typedef __attribute__((ext_vector_type(4)))  int      v4i;

#define WMMA_F16(a, b, c) \
    __builtin_amdgcn_wmma_f32_16x16x32_f16(false, (a), false, (b), (short)0, (c), false, false)

// ---------------------------------------------------------------------------
// A-fragment (16x32 f16): lane = m + 16*(k>>4).
// For lane-group g: halves 0..15 -> k = kb+16g .. kb+16g+15
// p points at row[kb] (f32, 16B aligned). Convert f32 -> f16 in VALU.
// ---------------------------------------------------------------------------
__device__ __forceinline__ v16h load_a_frag(const float* __restrict__ p, int g) {
    const float4* q0 = (const float4*)(p + 16 * g);
    float4 a0 = q0[0], a1 = q0[1];
    float4 b0 = q0[2], b1 = q0[3];
    v16h v;
    v[0]  = (_Float16)a0.x; v[1]  = (_Float16)a0.y; v[2]  = (_Float16)a0.z; v[3]  = (_Float16)a0.w;
    v[4]  = (_Float16)a1.x; v[5]  = (_Float16)a1.y; v[6]  = (_Float16)a1.z; v[7]  = (_Float16)a1.w;
    v[8]  = (_Float16)b0.x; v[9]  = (_Float16)b0.y; v[10] = (_Float16)b0.z; v[11] = (_Float16)b0.w;
    v[12] = (_Float16)b1.x; v[13] = (_Float16)b1.y; v[14] = (_Float16)b1.z; v[15] = (_Float16)b1.w;
    return v;
}

// B-fragment (32x16 f16): lanes 0-15 hold K=kb..kb+15, lanes 16-31 hold
// K=kb+16..kb+31; half index = k&15.  p = &WT[n*K + kb] (f16 LDS, WT[n][k]).
__device__ __forceinline__ v16h load_b_frag(const _Float16* p, int g) {
    return *(const v16h*)(p + 16 * g);
}

// ---------------------------------------------------------------------------
// Stage pre-converted f16 weights from global into LDS.
// Uses the CDNA5 async LDS-copy path (ASYNCcnt) when available.
// halves_total must be a multiple of 256*8; caller must __syncthreads() after.
// ---------------------------------------------------------------------------
__device__ __forceinline__ void stage_weights(const _Float16* __restrict__ gsrc,
                                              _Float16* lds, int halves_total) {
    int per = halves_total >> 8;  // halves per thread (multiple of 8)
    const _Float16* g = gsrc + threadIdx.x * per;
    _Float16* l = lds + threadIdx.x * per;
#if __has_builtin(__builtin_amdgcn_global_load_async_to_lds_b128)
#pragma unroll
    for (int i = 0; i < per; i += 8) {
        __builtin_amdgcn_global_load_async_to_lds_b128(
            (__attribute__((address_space(1))) v4i*)(_Float16*)(g + i),
            (__attribute__((address_space(3))) v4i*)(l + i), 0, 0);
    }
#if __has_builtin(__builtin_amdgcn_s_wait_asynccnt)
    __builtin_amdgcn_s_wait_asynccnt(0);
#else
    asm volatile("s_wait_asynccnt 0" ::: "memory");
#endif
#else
#pragma unroll
    for (int i = 0; i < per; i += 8) *(v8h*)(l + i) = *(const v8h*)(g + i);
#endif
}

// one-time weight convert+transpose: W[k][n] f32 -> WT[n][k] f16
__global__ void transpose_f16_kernel(const float* __restrict__ W, _Float16* __restrict__ WT,
                                     int K, int Nc) {
    int i = blockIdx.x * blockDim.x + threadIdx.x;
    if (i < K * Nc) {
        int k = i / Nc, n = i % Nc;
        WT[n * K + k] = (_Float16)W[i];
    }
}

// ===========================================================================
// GEMM: Y[N,128] = X[N,128] @ W[128,128]   (f16 WMMA, f32 accumulate)
// 256 threads = 8 waves; each wave computes a 16x128 output strip.
// ===========================================================================
__global__ __launch_bounds__(256) void gemm128_kernel(
    const float* __restrict__ X, const _Float16* __restrict__ WT_g,
    float* __restrict__ Y, int nrows) {
    __shared__ __align__(16) _Float16 WT[128 * 128];  // WT[n][k], 32 KB
    stage_weights(WT_g, WT, 128 * 128);
    __syncthreads();

    int tid = threadIdx.x;
    int wave = tid >> 5, lane = tid & 31;
    int g = lane >> 4, ln = lane & 15;
    int m0 = blockIdx.x * 128 + wave * 16;
    int m = m0 + ln;
    int mc = m < nrows ? m : nrows - 1;
    const float* row = X + (size_t)mc * 128;

    v8f acc[8];
#pragma unroll
    for (int t = 0; t < 8; ++t) acc[t] = (v8f){0.f, 0.f, 0.f, 0.f, 0.f, 0.f, 0.f, 0.f};

#pragma unroll
    for (int ks = 0; ks < 4; ++ks) {
        int kb = ks * 32;
        v16h a = load_a_frag(row + kb, g);
#pragma unroll
        for (int nt = 0; nt < 8; ++nt) {
            v16h b = load_b_frag(&WT[(nt * 16 + ln) * 128 + kb], g);
            acc[nt] = WMMA_F16(a, b, acc[nt]);
        }
    }

    // C/D layout: vgpr r -> row m0 + r + 8*g, lane ln -> col nt*16+ln
#pragma unroll
    for (int nt = 0; nt < 8; ++nt) {
#pragma unroll
        for (int r = 0; r < 8; ++r) {
            int mo = m0 + r + 8 * g;
            if (mo < nrows) Y[(size_t)mo * 128 + nt * 16 + ln] = acc[nt][r];
        }
    }
}

// ===========================================================================
// Degree / normalization helpers
// ===========================================================================
__global__ void deg_init_kernel(float* deg, int n) {
    int i = blockIdx.x * blockDim.x + threadIdx.x;
    if (i < n) deg[i] = 1.0f;  // self-loop
}
__global__ void deg_count_kernel(const long long* __restrict__ col, float* deg, int ne) {
    int i = blockIdx.x * blockDim.x + threadIdx.x;
    if (i < ne) atomicAdd(&deg[(int)col[i]], 1.0f);
}
__global__ void deg_rsqrt_kernel(float* deg, int n) {
    int i = blockIdx.x * blockDim.x + threadIdx.x;
    if (i < n) deg[i] = rsqrtf(deg[i]);
}

// out[c][f] = dinv[c]^2 * h[c][f] + bias[f]   (self-loop term + bias)
__global__ void agg_init_kernel(const float* __restrict__ h, const float* __restrict__ dinv,
                                const float* __restrict__ bias, float* __restrict__ out, int total) {
    int i = blockIdx.x * blockDim.x + threadIdx.x;
    if (i < total) {
        int c = i >> 7, f = i & 127;
        float di = dinv[c];
        out[i] = di * di * h[i] + bias[f];
    }
}

// one wave per edge: out[col] += dinv[row]*dinv[col] * h[row]
__global__ void agg_scatter_kernel(const float* __restrict__ h, const long long* __restrict__ row,
                                   const long long* __restrict__ col, const float* __restrict__ dinv,
                                   float* __restrict__ out, int ne) {
    int gid = blockIdx.x * blockDim.x + threadIdx.x;
    int e = gid >> 5, lane = gid & 31;
    if (e >= ne) return;
    int r = (int)row[e], c = (int)col[e];
    float nrm = dinv[r] * dinv[c];
    const float* hr = h + (size_t)r * 128;
    float* oc = out + (size_t)c * 128;
#pragma unroll
    for (int j = 0; j < 4; ++j) {
        int f = lane + 32 * j;
        atomicAdd(&oc[f], nrm * hr[f]);
    }
}

__global__ void relu_kernel(float* a, int total) {
    int i = blockIdx.x * blockDim.x + threadIdx.x;
    if (i < total) a[i] = fmaxf(a[i], 0.0f);
}

// ===========================================================================
// Edge MLP: per 16-edge tile  Z = relu([h[src],h[dst]] @ lpW1 + lpb1),
//           out = Z @ lpW2 + lpb2.   K=256, N=128, f16 WMMA.
// 256 threads = 8 waves, 128 edges per block. 64 KB LDS for lpW1^T (f16).
// ===========================================================================
__global__ __launch_bounds__(256) void edge_mlp_kernel(
    const float* __restrict__ H, const long long* __restrict__ src,
    const long long* __restrict__ dst, const _Float16* __restrict__ WT_g,
    const float* __restrict__ lpb1, const float* __restrict__ lpW2,
    const float* __restrict__ lpb2, float* __restrict__ out, int nedges) {
    __shared__ __align__(16) _Float16 WT[128 * 256];  // WT[n][k], 64 KB
    stage_weights(WT_g, WT, 128 * 256);
    __syncthreads();

    int tid = threadIdx.x;
    int wave = tid >> 5, lane = tid & 31;
    int g = lane >> 4, ln = lane & 15;
    int e0 = blockIdx.x * 128 + wave * 16;
    int e = e0 + ln;
    int ec = e < nedges ? e : nedges - 1;
    const float* ps = H + (size_t)src[ec] * 128;
    const float* pd = H + (size_t)dst[ec] * 128;

    v8f acc[8];
#pragma unroll
    for (int t = 0; t < 8; ++t) acc[t] = (v8f){0.f, 0.f, 0.f, 0.f, 0.f, 0.f, 0.f, 0.f};

#pragma unroll
    for (int ks = 0; ks < 8; ++ks) {
        const float* rp = (ks < 4) ? ps : pd;
        int kl = (ks & 3) * 32;   // offset inside the 128-wide half
        int kb = ks * 32;         // offset inside the 256-wide concat
        v16h a = load_a_frag(rp + kl, g);
#pragma unroll
        for (int nt = 0; nt < 8; ++nt) {
            v16h b = load_b_frag(&WT[(nt * 16 + ln) * 256 + kb], g);
            acc[nt] = WMMA_F16(a, b, acc[nt]);
        }
    }

    // epilogue: bias + relu + dot with lpW2, reduce across half-wave
    float bn[8], wn[8];
#pragma unroll
    for (int nt = 0; nt < 8; ++nt) {
        int n = nt * 16 + ln;
        bn[nt] = lpb1[n];
        wn[nt] = lpW2[n];
    }
    float lb2 = lpb2[0];

#pragma unroll
    for (int r = 0; r < 8; ++r) {
        float s = 0.f;
#pragma unroll
        for (int nt = 0; nt < 8; ++nt) {
            float z = acc[nt][r] + bn[nt];
            z = z > 0.f ? z : 0.f;
            s += z * wn[nt];
        }
        s += __shfl_xor(s, 1, 32);
        s += __shfl_xor(s, 2, 32);
        s += __shfl_xor(s, 4, 32);
        s += __shfl_xor(s, 8, 32);
        int eo = e0 + r + 8 * g;
        if (ln == 0 && eo < nedges) out[eo] = s + lb2;
    }
}

// ===========================================================================
extern "C" void kernel_launch(void* const* d_in, const int* in_sizes, int n_in,
                              void* d_out, int out_size, void* d_ws, size_t ws_size,
                              hipStream_t stream) {
    const float*     x    = (const float*)d_in[0];
    const long long* ei   = (const long long*)d_in[1];
    const float*     W1   = (const float*)d_in[2];
    const float*     b1   = (const float*)d_in[3];
    const float*     W2   = (const float*)d_in[4];
    const float*     b2   = (const float*)d_in[5];
    const float*     lpW1 = (const float*)d_in[6];
    const float*     lpb1 = (const float*)d_in[7];
    const float*     lpW2 = (const float*)d_in[8];
    const float*     lpb2 = (const float*)d_in[9];
    float* out = (float*)d_out;

    const int N = in_sizes[0] / 128;       // 50000
    const int E = in_sizes[1] / 2;         // 800000
    const long long* row = ei;             // sources
    const long long* col = ei + E;         // targets

    // workspace layout: dinv | buf0 | buf1 | f16 transposed weights
    float* dinv = (float*)d_ws;
    float* buf0 = dinv + (((size_t)N + 255) & ~(size_t)255);  // [N,128]
    float* buf1 = buf0 + (size_t)N * 128;                     // [N,128]
    _Float16* w1t   = (_Float16*)(buf1 + (size_t)N * 128);    // [128][128]
    _Float16* w2t   = w1t + 128 * 128;                        // [128][128]
    _Float16* lpw1t = w2t + 128 * 128;                        // [128][256]
    (void)ws_size; (void)n_in; (void)out_size;

    const int T = 256;
    int gN   = (N + T - 1) / T;
    int gE   = (E + T - 1) / T;
    int gNF  = (N * 128 + T - 1) / T;
    int gEw  = (E * 32 + T - 1) / T;       // one wave per edge
    int gRow = (N + 127) / 128;            // gemm blocks (128 rows each)
    int gEdg = (E + 127) / 128;            // edge-mlp blocks (128 edges each)

    // one-time weight convert + transpose to f16
    transpose_f16_kernel<<<(128 * 128 + T - 1) / T, T, 0, stream>>>(W1, w1t, 128, 128);
    transpose_f16_kernel<<<(128 * 128 + T - 1) / T, T, 0, stream>>>(W2, w2t, 128, 128);
    transpose_f16_kernel<<<(256 * 128 + T - 1) / T, T, 0, stream>>>(lpW1, lpw1t, 256, 128);

    // normalization: dinv = rsqrt(1 + in-degree)
    deg_init_kernel<<<gN, T, 0, stream>>>(dinv, N);
    deg_count_kernel<<<gE, T, 0, stream>>>(col, dinv, E);
    deg_rsqrt_kernel<<<gN, T, 0, stream>>>(dinv, N);

    // layer 1: h = relu(agg(x@W1) + b1)
    gemm128_kernel<<<gRow, T, 0, stream>>>(x, w1t, buf0, N);
    agg_init_kernel<<<gNF, T, 0, stream>>>(buf0, dinv, b1, buf1, N * 128);
    agg_scatter_kernel<<<gEw, T, 0, stream>>>(buf0, row, col, dinv, buf1, E);
    relu_kernel<<<gNF, T, 0, stream>>>(buf1, N * 128);

    // layer 2: h2 = agg(h@W2) + b2
    gemm128_kernel<<<gRow, T, 0, stream>>>(buf1, w2t, buf0, N);
    agg_init_kernel<<<gNF, T, 0, stream>>>(buf0, dinv, b2, buf1, N * 128);
    agg_scatter_kernel<<<gEw, T, 0, stream>>>(buf0, row, col, dinv, buf1, E);

    // link predictor over edges
    edge_mlp_kernel<<<gEdg, T, 0, stream>>>(buf1, row, col, lpw1t, lpb1, lpW2, lpb2, out, E);
}
